// _GDNAttention_66855460929594
// MI455X (gfx1250) — matrix operand order repaired and checked by reading, not compiled, not run on hardware
//
#include <hip/hip_runtime.h>
#include <hip/hip_bf16.h>

// ---------------------------------------------------------------------------
// Gated DeltaNet block for MI455X (gfx1250, wave32, WMMA).
//   1) cast x -> bf16; transpose+cast weights -> bf16 [N][K]
//   2) WMMA bf16 GEMMs (128x64 block tile, 32x32 wave tile, double-buffered
//      LDS fed by GLOBAL_LOAD_ASYNC_TO_LDS_B128 when available)
//   3) beta / log-decay g kernel
//   4) causal conv4 + SiLU (+ per-head l2norm for q,k) -> bf16 [B,H,T,*]
//   5) chunked gated delta rule (chunk=16) — all chunk matmuls on WMMA,
//      state S (128x256) kept transposed in LDS as bf16 [256][136]
//   6) gated RMSNorm * silu(gate) -> bf16
//   7) WMMA GEMM o @ Wo -> d_out (f32)
// ---------------------------------------------------------------------------

#define B_  8
#define T_  2048
#define D_  768
#define H_  6
#define DK_ 128
#define DV_ 256
#define KD_ 768      // H*DK
#define VD_ 1536     // H*DV
#define BT_ (B_ * T_)
#define CH_ 16       // delta-rule chunk length

typedef __bf16 bf16_t;
typedef bf16_t bf16x16 __attribute__((ext_vector_type(16)));
typedef float  floatx8 __attribute__((ext_vector_type(8)));
typedef int    v4i __attribute__((vector_size(4 * sizeof(int))));  // matches builtin param

__device__ __forceinline__ float siluf(float x) { return x / (1.0f + __expf(-x)); }

// ---- CDNA5 async global->LDS copy (ASYNCcnt) with sync fallback -----------
#if __has_builtin(__builtin_amdgcn_global_load_async_to_lds_b128) && \
    __has_builtin(__builtin_amdgcn_s_wait_asynccnt)
#define USE_ASYNC_LDS 1
#else
#define USE_ASYNC_LDS 0
#endif

__device__ __forceinline__ void copy_b128_g2l(const bf16_t* g, bf16_t* l) {
#if USE_ASYNC_LDS
  __builtin_amdgcn_global_load_async_to_lds_b128((v4i*)g, (v4i*)l, 0, 0);
#else
  *(uint4*)l = *(const uint4*)g;
#endif
}
__device__ __forceinline__ void async_copy_wait() {
#if USE_ASYNC_LDS
  __builtin_amdgcn_s_wait_asynccnt(0);
#endif
}

// ---- WMMA fragment loaders (bf16, 16x16x32) -------------------------------
// A-matrix (16x32, row-major in LDS, ldm elements): lane = half*16 + m
//   elems 0..7  -> K = k0 + half*8 + j ; elems 8..15 -> K = k0 + 16 + half*8 + j
__device__ __forceinline__ bf16x16 frag_A(const bf16_t* base, int row0, int ldm, int k0) {
  int lane = threadIdx.x & 31;
  const bf16_t* p = base + (size_t)(row0 + (lane & 15)) * ldm + k0 + ((lane >> 4) << 3);
  bf16x16 f;
#pragma unroll
  for (int j = 0; j < 8; ++j) { f[j] = p[j]; f[j + 8] = p[j + 16]; }
  return f;
}
// B-matrix (32x16): stored K-contiguous per column: row `col` of base holds B[:,col]
//   lane = half*16 + n ; elems 0..7 -> K = k0 + half*8 + j ; elems 8..15 -> K = k0 + 16 + half*8 + j
__device__ __forceinline__ bf16x16 frag_B(const bf16_t* base, int col0, int ldm, int k0) {
  int lane = threadIdx.x & 31;
  const bf16_t* p = base + (size_t)(col0 + (lane & 15)) * ldm + k0 + ((lane >> 4) << 3);
  bf16x16 f;
#pragma unroll
  for (int j = 0; j < 8; ++j) { f[j] = p[j]; f[j + 8] = p[j + 16]; }
  return f;
}

#define WMMA_BF16(A, Bf, C) \
  __builtin_amdgcn_wmma_f32_16x16x32_bf16(false, (A), false, (Bf), (short)0, (C), false, false)

// ---------------------------------------------------------------------------
__global__ __launch_bounds__(256) void cast_bf16_kernel(const float* __restrict__ src,
                                                        bf16_t* __restrict__ dst, int n) {
  int i = blockIdx.x * 256 + threadIdx.x;
  if (i < n) dst[i] = (bf16_t)src[i];
}

// W[K][N] f32 -> Wt[N][K] bf16
__global__ __launch_bounds__(256) void transpose_cast_kernel(const float* __restrict__ W,
                                                             bf16_t* __restrict__ Wt,
                                                             int K, int N) {
  int i = blockIdx.x * 256 + threadIdx.x;
  if (i >= K * N) return;
  int n = i / K, k = i % K;
  Wt[i] = (bf16_t)W[(size_t)k * N + n];
}

// ---------------------------------------------------------------------------
// C[M][N] f32 = A[M][K] bf16 @ W (given as Wt[N][K] bf16).
// Block tile 128x64 (8 waves, 4x2 grid of 32x32 wave tiles, 4 WMMA accs/wave).
// Double-buffered LDS; tiles staged with async global->LDS DMA.
// Requires M%128==0, N%64==0, K%32==0.
__global__ __launch_bounds__(256) void gemm_bf16_wmma(const bf16_t* __restrict__ A,
                                                      const bf16_t* __restrict__ Bt,
                                                      float* __restrict__ C,
                                                      int M, int N, int K) {
  __shared__ bf16_t sA[2][128 * 40];
  __shared__ bf16_t sB[2][64 * 40];
  const int tid  = threadIdx.x;
  const int wave = tid >> 5, lane = tid & 31;
  const int m0 = blockIdx.y * 128, n0 = blockIdx.x * 64;
  const int wm = (wave >> 1) << 5;   // 0,32,64,96
  const int wn = (wave & 1) << 5;    // 0,32
  floatx8 acc00 = {}, acc01 = {}, acc10 = {}, acc11 = {};

  auto load_tiles = [&](int buf, int k0) {
#pragma unroll
    for (int u = 0; u < 2; ++u) {        // A: 128x32 = 512 b128 chunks
      int idx = tid + (u << 8);
      int r = idx >> 2, c = (idx & 3) << 3;
      copy_b128_g2l(&A[(size_t)(m0 + r) * K + k0 + c], &sA[buf][r * 40 + c]);
    }
    {                                    // B: 64x32 = 256 b128 chunks
      int r = tid >> 2, c = (tid & 3) << 3;
      copy_b128_g2l(&Bt[(size_t)(n0 + r) * K + k0 + c], &sB[buf][r * 40 + c]);
    }
  };

  load_tiles(0, 0);
  async_copy_wait();
  __syncthreads();

  int buf = 0;
  for (int k0 = 0; k0 < K; k0 += 32) {
    if (k0 + 32 < K) load_tiles(buf ^ 1, k0 + 32);
    bf16x16 a0 = frag_A(sA[buf], wm, 40, 0);
    bf16x16 a1 = frag_A(sA[buf], wm + 16, 40, 0);
    bf16x16 b0 = frag_B(sB[buf], wn, 40, 0);
    bf16x16 b1 = frag_B(sB[buf], wn + 16, 40, 0);
    acc00 = WMMA_BF16(a0, b0, acc00);
    acc01 = WMMA_BF16(a0, b1, acc01);
    acc10 = WMMA_BF16(a1, b0, acc10);
    acc11 = WMMA_BF16(a1, b1, acc11);
    async_copy_wait();
    __syncthreads();
    buf ^= 1;
  }

  const int n = lane & 15, half = lane >> 4;
#pragma unroll
  for (int r = 0; r < 8; ++r) {
    int row = m0 + wm + (half ? r + 8 : r);
    C[(size_t)row * N + n0 + wn + n]             = acc00[r];
    C[(size_t)row * N + n0 + wn + 16 + n]        = acc01[r];
    C[(size_t)(row + 16) * N + n0 + wn + n]      = acc10[r];
    C[(size_t)(row + 16) * N + n0 + wn + 16 + n] = acc11[r];
  }
}

// ---------------------------------------------------------------------------
// beta = sigmoid(x@Wb), g = -exp(A_log)*softplus(x@Wa + dt_bias); layout [B,H,T]
__global__ __launch_bounds__(256) void ab_kernel(const float* __restrict__ x,
                                                 const float* __restrict__ Wa,
                                                 const float* __restrict__ Wb,
                                                 const float* __restrict__ A_log,
                                                 const float* __restrict__ dt_bias,
                                                 float* __restrict__ gdec,
                                                 float* __restrict__ beta) {
  int idx = blockIdx.x * 256 + threadIdx.x;
  if (idx >= BT_ * H_) return;
  int h = idx % H_, bt = idx / H_;
  const float* xp = x + (size_t)bt * D_;
  float sa = 0.f, sb = 0.f;
  for (int d = 0; d < D_; ++d) {
    float xv = xp[d];
    sa += xv * Wa[d * H_ + h];
    sb += xv * Wb[d * H_ + h];
  }
  float bta = 1.0f / (1.0f + __expf(-sb));
  float z = sa + dt_bias[h];
  float sp = (z > 20.f) ? z : log1pf(__expf(z));
  float g = -__expf(A_log[h]) * sp;
  int t = bt % T_, b = bt / T_;
  size_t o = (size_t)(b * H_ + h) * T_ + t;
  gdec[o] = g;
  beta[o] = bta;
}

// ---------------------------------------------------------------------------
// causal conv4 + SiLU + per-head l2norm; one wave per (b,t,h); lane owns 4 chans
__global__ __launch_bounds__(256) void convqk_kernel(const float* __restrict__ proj,
                                                     const float* __restrict__ cw,
                                                     bf16_t* __restrict__ outp) {
  int gw = blockIdx.x * 8 + (threadIdx.x >> 5);
  if (gw >= BT_ * H_) return;
  int lane = threadIdx.x & 31;
  int h = gw % H_, bt = gw / H_;
  int t = bt % T_, b = bt / T_;
  float vals[4]; float ss = 0.f;
#pragma unroll
  for (int j = 0; j < 4; ++j) {
    int c = h * DK_ + lane * 4 + j;
    float acc = 0.f;
#pragma unroll
    for (int i = 0; i < 4; ++i) {
      int tt = t - 3 + i;
      if (tt >= 0) acc += proj[((size_t)b * T_ + tt) * KD_ + c] * cw[c * 4 + i];
    }
    acc = siluf(acc);
    vals[j] = acc; ss += acc * acc;
  }
#pragma unroll
  for (int off = 16; off > 0; off >>= 1) ss += __shfl_xor(ss, off, 32);
  float inv = rsqrtf(ss + 1e-6f);
#pragma unroll
  for (int j = 0; j < 4; ++j)
    outp[((size_t)(b * H_ + h) * T_ + t) * DK_ + lane * 4 + j] = (bf16_t)(vals[j] * inv);
}

// causal conv4 + SiLU for v -> bf16 [B,H,T,DV]
__global__ __launch_bounds__(256) void convv_kernel(const float* __restrict__ proj,
                                                    const float* __restrict__ cw,
                                                    bf16_t* __restrict__ outp) {
  size_t idx = (size_t)blockIdx.x * 256 + threadIdx.x;
  if (idx >= (size_t)BT_ * VD_) return;
  int c = (int)(idx % VD_);
  size_t bt = idx / VD_;
  int t = (int)(bt % T_), b = (int)(bt / T_);
  float acc = 0.f;
#pragma unroll
  for (int i = 0; i < 4; ++i) {
    int tt = t - 3 + i;
    if (tt >= 0) acc += proj[((size_t)b * T_ + tt) * VD_ + c] * cw[c * 4 + i];
  }
  acc = siluf(acc);
  int h = c >> 8, dv = c & 255;
  outp[((size_t)(b * H_ + h) * T_ + t) * DV_ + dv] = (bf16_t)acc;
}

// ---------------------------------------------------------------------------
// Chunked gated delta rule. One workgroup (256 thr = 8 waves) per (b,h).
// State S^T kept in LDS as bf16 [DV][136]. Chunk = 16 timesteps.
//   lam_t = exp(cum g), w_t = beta_t*lam_t*k_t, khat_t = exp(ctot-cum_t)*k_t
//   r = beta*v - w@S ; (I + tril(A)) u = r, A[i][j]=beta_i lam_i/lam_j (k_i.k_j)
//   o = (scale*lam*q)@S + AttM@u ; S^T = exp(ctot)*S^T + U^T @ Khat
#define DELTA_LDS_BF16 (256*136 + 3*16*136 + 128*40 + 256*40 + 16*40)
#define DELTA_LDS_F32  (16*256*2 + 256*3 + 65)
#define DELTA_SMEM     (DELTA_LDS_BF16*2 + DELTA_LDS_F32*4)

__global__ __launch_bounds__(256) void delta_kernel(const bf16_t* __restrict__ q_s,
                                                    const bf16_t* __restrict__ k_s,
                                                    const bf16_t* __restrict__ v_s,
                                                    const float* __restrict__ gdec,
                                                    const float* __restrict__ beta,
                                                    float* __restrict__ o) {
  extern __shared__ char smem_raw[];
  bf16_t* S    = (bf16_t*)smem_raw;        // [256][136]  S^T (bf16 master state)
  bf16_t* kcb  = S    + 256 * 136;         // [16][136]   k chunk
  bf16_t* wcb  = kcb  + 16 * 136;          // [16][136]   beta*lam*k
  bf16_t* qcb  = wcb  + 16 * 136;          // [16][136]   scale*lam*q
  bf16_t* khT  = qcb  + 16 * 136;          // [128][40]   khat^T (cols 16..31 = 0)
  bf16_t* Ub   = khT  + 128 * 40;          // [256][40]   u^T    (cols 16..31 = 0)
  bf16_t* AttM = Ub   + 256 * 40;          // [16][40]    causal attn (cols 16..31 = 0)
  float*  ru   = (float*)(AttM + 16 * 40); // [16][256]   r -> u
  float*  oac  = ru   + 16 * 256;          // [16][256]   inter-chunk output
  float*  sck  = oac  + 16 * 256;          // [16][16]    k.k^T
  float*  scq  = sck  + 256;               // [16][16]    (scale*lam*q).k^T
  float*  amat = scq  + 256;               // [16][16]
  float*  cumg = amat + 256;               // [16]
  float*  lam  = cumg + 16;
  float*  bet  = lam  + 16;
  float*  khs  = bet  + 16;
  float*  ct   = khs  + 16;                // [1]

  const int bh = blockIdx.x;
  const int b = bh / H_, h = bh % H_;
  const int tid = threadIdx.x, lane = tid & 31, wave = tid >> 5;
  const float SCALE = 0.08838834764831845f;  // DK^-0.5
  const size_t bhT = (size_t)(b * H_ + h) * T_;

  for (int i = tid; i < 256 * 136; i += 256) S[i]    = (bf16_t)0.0f;
  for (int i = tid; i < 128 * 40;  i += 256) khT[i]  = (bf16_t)0.0f;
  for (int i = tid; i < 256 * 40;  i += 256) Ub[i]   = (bf16_t)0.0f;
  for (int i = tid; i < 16 * 40;   i += 256) AttM[i] = (bf16_t)0.0f;
  __syncthreads();

  for (int ch = 0; ch < T_ / CH_; ++ch) {
    const int tb = ch * CH_;
    if (tid == 0) {
      float c = 0.f;
      for (int i = 0; i < CH_; ++i) { c += gdec[bhT + tb + i]; cumg[i] = c; }
      ct[0] = c;
    }
    __syncthreads();
    if (tid < CH_) {
      lam[tid] = __expf(cumg[tid]);
      khs[tid] = __expf(ct[0] - cumg[tid]);
      bet[tid] = beta[bhT + tb + tid];
    }
    __syncthreads();
    const float ectot = __expf(ct[0]);

    {  // load q/k chunk with scalings folded; 8 elems/thread
      int base = tid * 8;
      int i = base >> 7, m = base & 127;
      const bf16_t* kp = k_s + (bhT + tb + i) * DK_ + m;
      const bf16_t* qp = q_s + (bhT + tb + i) * DK_ + m;
      float li = lam[i], bi = bet[i], ki = khs[i];
#pragma unroll
      for (int j = 0; j < 8; ++j) {
        float kv = (float)kp[j];
        float qv = (float)qp[j];
        kcb[i * 136 + m + j]  = (bf16_t)kv;
        wcb[i * 136 + m + j]  = (bf16_t)(bi * li * kv);
        qcb[i * 136 + m + j]  = (bf16_t)(SCALE * li * qv);
        khT[(m + j) * 40 + i] = (bf16_t)(ki * kv);
      }
    }
    {  // r = beta * v ; 16 elems/thread
      int base = tid * 16;
      int i = base >> 8, n = base & 255;
      const bf16_t* vp = v_s + (bhT + tb + i) * DV_ + n;
      float bi = bet[i];
#pragma unroll
      for (int j = 0; j < 16; ++j) ru[i * 256 + n + j] = bi * (float)vp[j];
    }
    __syncthreads();

    // GEMM phase: r -= w@S ; oac = (scale*lam*q)@S ; score matrices
#pragma unroll
    for (int s = 0; s < 2; ++s) {
      int nt = wave * 2 + s;
      floatx8 a1 = {}, a2 = {};
#pragma unroll
      for (int kk = 0; kk < 4; ++kk) {
        bf16x16 bf = frag_B(S, nt * 16, 136, kk * 32);
        bf16x16 aw = frag_A(wcb, 0, 136, kk * 32);
        bf16x16 aq = frag_A(qcb, 0, 136, kk * 32);
        a1 = WMMA_BF16(aw, bf, a1);
        a2 = WMMA_BF16(aq, bf, a2);
      }
      int n = lane & 15, half = lane >> 4;
#pragma unroll
      for (int r = 0; r < 8; ++r) {
        int row = half ? r + 8 : r;
        ru[row * 256 + nt * 16 + n]  -= a1[r];
        oac[row * 256 + nt * 16 + n]  = a2[r];
      }
    }
    if (wave == 0) {  // sck = k @ k^T
      floatx8 a = {};
#pragma unroll
      for (int kk = 0; kk < 4; ++kk)
        a = WMMA_BF16(frag_A(kcb, 0, 136, kk * 32), frag_B(kcb, 0, 136, kk * 32), a);
      int n = lane & 15, half = lane >> 4;
#pragma unroll
      for (int r = 0; r < 8; ++r) sck[(half ? r + 8 : r) * 16 + n] = a[r];
    }
    if (wave == 1) {  // scq = (scale*lam*q) @ k^T
      floatx8 a = {};
#pragma unroll
      for (int kk = 0; kk < 4; ++kk)
        a = WMMA_BF16(frag_A(qcb, 0, 136, kk * 32), frag_B(kcb, 0, 136, kk * 32), a);
      int n = lane & 15, half = lane >> 4;
#pragma unroll
      for (int r = 0; r < 8; ++r) scq[(half ? r + 8 : r) * 16 + n] = a[r];
    }
    __syncthreads();

    {  // build AttM (causal, incl diag) and strictly-lower amat
      int i = tid >> 4, j = tid & 15;
      float inv = 1.0f / lam[j];
      AttM[i * 40 + j] = (bf16_t)((j <= i) ? scq[i * 16 + j] * inv : 0.0f);
      amat[i * 16 + j] = (j < i) ? bet[i] * lam[i] * sck[i * 16 + j] * inv : 0.0f;
    }
    __syncthreads();

    {  // forward substitution: (I + tril(amat)) u = r ; thread owns column n
      int n = tid;
      float u[CH_];
#pragma unroll
      for (int i = 0; i < CH_; ++i) {
        float val = ru[i * 256 + n];
        for (int j = 0; j < i; ++j) val -= amat[i * 16 + j] * u[j];
        u[i] = val;
        Ub[n * 40 + i] = (bf16_t)val;
      }
    }
    __syncthreads();

    // intra-chunk output: o = oac + AttM @ U ; store to global
#pragma unroll
    for (int s = 0; s < 2; ++s) {
      int nt = wave * 2 + s;
      bf16x16 a   = frag_A(AttM, 0, 40, 0);
      bf16x16 bfr = frag_B(Ub, nt * 16, 40, 0);
      floatx8 c;
      int n = lane & 15, half = lane >> 4;
#pragma unroll
      for (int r = 0; r < 8; ++r) c[r] = oac[(half ? r + 8 : r) * 256 + nt * 16 + n];
      c = WMMA_BF16(a, bfr, c);
#pragma unroll
      for (int r = 0; r < 8; ++r) {
        int row = half ? r + 8 : r;
        o[(((size_t)b * T_ + tb + row) * H_ + h) * DV_ + nt * 16 + n] = c[r];
      }
    }

    // state update: S^T = exp(ctot)*S^T + U^T @ Khat  (128 tiles, 16 per wave)
    for (int tt = 0; tt < 16; ++tt) {
      int tile = wave * 16 + tt;
      int mt = tile >> 3;   // dv tile 0..15
      int nt = tile & 7;    // dk tile 0..7
      bf16x16 a   = frag_A(Ub, mt * 16, 40, 0);
      bf16x16 bfr = frag_B(khT, nt * 16, 40, 0);
      floatx8 c;
      int n = lane & 15, half = lane >> 4;
#pragma unroll
      for (int r = 0; r < 8; ++r) {
        int row = mt * 16 + (half ? r + 8 : r);
        c[r] = ectot * (float)S[row * 136 + nt * 16 + n];
      }
      c = WMMA_BF16(a, bfr, c);
#pragma unroll
      for (int r = 0; r < 8; ++r) {
        int row = mt * 16 + (half ? r + 8 : r);
        S[row * 136 + nt * 16 + n] = (bf16_t)c[r];
      }
    }
    __syncthreads();
  }
}

// ---------------------------------------------------------------------------
// gated RMSNorm * silu(gate) -> bf16 ; one wave per (b,t,h), lane owns 8 dv
__global__ __launch_bounds__(256) void gatenorm_kernel(const float* __restrict__ o,
                                                       const float* __restrict__ gproj,
                                                       const float* __restrict__ onw,
                                                       bf16_t* __restrict__ ob) {
  int gw = blockIdx.x * 8 + (threadIdx.x >> 5);
  if (gw >= BT_ * H_) return;
  int lane = threadIdx.x & 31;
  int h = gw % H_;
  size_t bt = gw / H_;
  const float* op = o + (bt * H_ + h) * DV_;
  float v[8]; float ss = 0.f;
#pragma unroll
  for (int j = 0; j < 8; ++j) { v[j] = op[lane * 8 + j]; ss += v[j] * v[j]; }
#pragma unroll
  for (int off = 16; off > 0; off >>= 1) ss += __shfl_xor(ss, off, 32);
  float inv = rsqrtf(ss * (1.0f / DV_) + 1e-5f);
#pragma unroll
  for (int j = 0; j < 8; ++j) {
    int dv = lane * 8 + j;
    float gt = gproj[bt * VD_ + h * DV_ + dv];
    ob[bt * VD_ + h * DV_ + dv] = (bf16_t)(v[j] * inv * onw[dv] * siluf(gt));
  }
}

// ---------------------------------------------------------------------------
extern "C" void kernel_launch(void* const* d_in, const int* in_sizes, int n_in,
                              void* d_out, int out_size, void* d_ws, size_t ws_size,
                              hipStream_t stream) {
  const float* x       = (const float*)d_in[0];
  const float* Wq      = (const float*)d_in[1];
  const float* Wk      = (const float*)d_in[2];
  const float* Wv      = (const float*)d_in[3];
  const float* Wa      = (const float*)d_in[4];
  const float* Wb      = (const float*)d_in[5];
  const float* Wg      = (const float*)d_in[6];
  const float* Wo      = (const float*)d_in[7];
  const float* conv_q  = (const float*)d_in[8];
  const float* conv_k  = (const float*)d_in[9];
  const float* conv_v  = (const float*)d_in[10];
  const float* A_log   = (const float*)d_in[11];
  const float* dt_bias = (const float*)d_in[12];
  const float* onw     = (const float*)d_in[13];
  float* out = (float*)d_out;

  char* w = (char*)d_ws;
  size_t off = 0;
  auto alloc = [&](size_t bytes) -> void* {
    void* p = w + off;
    off = (off + bytes + 255) & ~(size_t)255;
    return p;
  };
  bf16_t* xbf   = (bf16_t*)alloc((size_t)BT_ * D_ * 2);
  bf16_t* WqT   = (bf16_t*)alloc((size_t)KD_ * D_ * 2);
  bf16_t* WkT   = (bf16_t*)alloc((size_t)KD_ * D_ * 2);
  bf16_t* WvT   = (bf16_t*)alloc((size_t)VD_ * D_ * 2);
  bf16_t* WgT   = (bf16_t*)alloc((size_t)VD_ * D_ * 2);
  bf16_t* WoT   = (bf16_t*)alloc((size_t)D_ * VD_ * 2);
  float*  qproj = (float*)alloc((size_t)BT_ * KD_ * 4);
  float*  kproj = (float*)alloc((size_t)BT_ * KD_ * 4);
  float*  vproj = (float*)alloc((size_t)BT_ * VD_ * 4);
  float*  gproj = (float*)alloc((size_t)BT_ * VD_ * 4);
  bf16_t* q_s   = (bf16_t*)alloc((size_t)BT_ * KD_ * 2);
  bf16_t* k_s   = (bf16_t*)alloc((size_t)BT_ * KD_ * 2);
  bf16_t* v_s   = (bf16_t*)alloc((size_t)BT_ * VD_ * 2);
  float*  gdec  = (float*)alloc((size_t)B_ * H_ * T_ * 4);
  float*  beta  = (float*)alloc((size_t)B_ * H_ * T_ * 4);
  float*  o_att = (float*)alloc((size_t)BT_ * VD_ * 4);
  bf16_t* ob    = (bf16_t*)alloc((size_t)BT_ * VD_ * 2);
  (void)ws_size; (void)n_in; (void)in_sizes; (void)out_size;

  // 1) casts / transposes
  {
    int n = BT_ * D_;
    cast_bf16_kernel<<<(n + 255) / 256, 256, 0, stream>>>(x, xbf, n);
    int nw = D_ * KD_;
    transpose_cast_kernel<<<(nw + 255) / 256, 256, 0, stream>>>(Wq, WqT, D_, KD_);
    transpose_cast_kernel<<<(nw + 255) / 256, 256, 0, stream>>>(Wk, WkT, D_, KD_);
    int nv = D_ * VD_;
    transpose_cast_kernel<<<(nv + 255) / 256, 256, 0, stream>>>(Wv, WvT, D_, VD_);
    transpose_cast_kernel<<<(nv + 255) / 256, 256, 0, stream>>>(Wg, WgT, D_, VD_);
    transpose_cast_kernel<<<(nv + 255) / 256, 256, 0, stream>>>(Wo, WoT, VD_, D_);
  }
  // 2) projection GEMMs (128x64 block tiles)
  {
    dim3 gq(KD_ / 64, BT_ / 128);
    gemm_bf16_wmma<<<gq, 256, 0, stream>>>(xbf, WqT, qproj, BT_, KD_, D_);
    gemm_bf16_wmma<<<gq, 256, 0, stream>>>(xbf, WkT, kproj, BT_, KD_, D_);
    dim3 gv(VD_ / 64, BT_ / 128);
    gemm_bf16_wmma<<<gv, 256, 0, stream>>>(xbf, WvT, vproj, BT_, VD_, D_);
    gemm_bf16_wmma<<<gv, 256, 0, stream>>>(xbf, WgT, gproj, BT_, VD_, D_);
  }
  // 3) beta / decay
  ab_kernel<<<(BT_ * H_ + 255) / 256, 256, 0, stream>>>(x, Wa, Wb, A_log, dt_bias, gdec, beta);
  // 4) conv + silu (+ l2norm)
  {
    int blocks = (BT_ * H_) / 8;
    convqk_kernel<<<blocks, 256, 0, stream>>>(qproj, conv_q, q_s);
    convqk_kernel<<<blocks, 256, 0, stream>>>(kproj, conv_k, k_s);
    size_t nv = (size_t)BT_ * VD_;
    convv_kernel<<<(unsigned)((nv + 255) / 256), 256, 0, stream>>>(vproj, conv_v, v_s);
  }
  // 5) chunked gated delta rule (48 workgroups, ~147 KB LDS each)
  {
    (void)hipFuncSetAttribute((const void*)delta_kernel,
                              hipFuncAttributeMaxDynamicSharedMemorySize, DELTA_SMEM);
    delta_kernel<<<B_ * H_, 256, DELTA_SMEM, stream>>>(q_s, k_s, v_s, gdec, beta, o_att);
  }
  // 6) gated RMSNorm * silu(gate)
  gatenorm_kernel<<<(BT_ * H_) / 8, 256, 0, stream>>>(o_att, gproj, onw, ob);
  // 7) output projection
  {
    dim3 go(D_ / 64, BT_ / 128);
    gemm_bf16_wmma<<<go, 256, 0, stream>>>(ob, WoT, out, BT_, D_, VD_);
  }
}
